// GCNN_45896020525657
// MI455X (gfx1250) — matrix operand order repaired and checked by reading, not compiled, not run on hardware
//
#include <hip/hip_runtime.h>
#include <hip/hip_bf16.h>
#include <stdint.h>

// ---------------------------------------------------------------------------
// GCNN on MI455X (gfx1250, wave32).
// Core: out[b,g,d] = gelu( scale * sum_{h,c} x[b,h,c] * K[T[g,h],c,d] )
// mapped to v_wmma_f32_16x16x32_bf16:  M=16 batch rows, N=16 = 2 g x 8 d,
// K=32 = 4 h x 8 c.  A-panel (16 x 4096 bf16 = 128KB) staged in LDS via
// async-to-LDS and reused across all g; B streamed from a precomputed bf16
// tensor Kg[g][h][d][c] resident in L2 (33.5MB << 192MB).
// ---------------------------------------------------------------------------

typedef __attribute__((ext_vector_type(16))) __bf16 v16bf;
typedef __attribute__((ext_vector_type(8)))  __bf16 v8bf;
typedef __attribute__((ext_vector_type(8)))  float  v8f;
typedef __attribute__((ext_vector_type(4)))  int    v4i;

#define GG     512          // group order
#define CF     8            // feature channels
#define KDIM   4096         // GG*CF contraction length
#define NBATCH 1024
#define AROW_BYTES (KDIM * 2 + 16)   // 8208 B: +16B pad -> 4-bank shift/row

#define TRY_ASYNC 1
#if TRY_ASYNC && defined(__has_builtin)
#if __has_builtin(__builtin_amdgcn_global_load_async_to_lds_b128) && \
    __has_builtin(__builtin_amdgcn_s_wait_asynccnt)
#define USE_ASYNC 1
#endif
#endif
#ifndef USE_ASYNC
#define USE_ASYNC 0
#endif

// Branch-free tanh: prefer gfx1250 hardware v_tanh_f32, else exp2-based.
__device__ __forceinline__ float fast_tanh(float x) {
#if defined(__has_builtin) && __has_builtin(__builtin_amdgcn_tanhf)
  return __builtin_amdgcn_tanhf(x);
#elif defined(__has_builtin) && __has_builtin(__builtin_amdgcn_exp2f) && \
      __has_builtin(__builtin_amdgcn_rcpf)
  // tanh(x) = 1 - 2/(exp(2x)+1);  exp(2x) = exp2(2*log2(e)*x)
  float e = __builtin_amdgcn_exp2f(2.8853900817779268f * x);
  return 1.0f - 2.0f * __builtin_amdgcn_rcpf(e + 1.0f);
#else
  float e = __expf(2.0f * x);
  return 1.0f - 2.0f / (e + 1.0f);
#endif
}

__device__ __forceinline__ float gelu_tanh(float x) {
  // jax.nn.gelu default (approximate=True)
  float x3 = x * x * x;
  return 0.5f * x * (1.0f + fast_tanh(0.7978845608028654f * (x + 0.044715f * x3)));
}

// --------------------------------------------------------------------------
// Lift: x0[b][h*8+c] = gelu( sum_n s[b, perms[h,n]] * W_lift[n,c] + b_lift[c] )
// grid (512, 4) x 256 threads; one thread = one (b,h) pair (8 outputs).
// --------------------------------------------------------------------------
__global__ __launch_bounds__(256) void lift_kernel(
    const float* __restrict__ s, const int* __restrict__ perms,
    const float* __restrict__ Wl, const float* __restrict__ bl,
    __bf16* __restrict__ x0) {
  __shared__ int   pr[64];
  __shared__ float W[64 * 8];
  __shared__ float bb[8];
  const int h = blockIdx.x;
  const int tid = threadIdx.x;
  if (tid < 64) pr[tid] = perms[h * 64 + tid];
  for (int i = tid; i < 512; i += 256) W[i] = Wl[i];
  if (tid < 8) bb[tid] = bl[tid];
  __syncthreads();
  const int b = blockIdx.y * 256 + tid;
  float acc[8];
#pragma unroll
  for (int c = 0; c < 8; ++c) acc[c] = bb[c];
  const float* srow = s + (size_t)b * 64;
  for (int n = 0; n < 64; ++n) {
    float sv = srow[pr[n]];
#pragma unroll
    for (int c = 0; c < 8; ++c) acc[c] += sv * W[n * 8 + c];
  }
  __bf16* dst = x0 + (size_t)b * KDIM + h * 8;
#pragma unroll
  for (int c = 0; c < 8; ++c) dst[c] = (__bf16)gelu_tanh(acc[c]);
}

// --------------------------------------------------------------------------
// Pack: Kg[g][h][d][c] = bf16( K[ mult[g, inv[h]] ][c][d] )
// one thread per (g,h); reads one contiguous 256B K row, writes 128B.
// --------------------------------------------------------------------------
__global__ __launch_bounds__(256) void pack_kernel(
    const float* __restrict__ K, const int* __restrict__ mult,
    const int* __restrict__ inv, __bf16* __restrict__ kg) {
  const int idx = blockIdx.x * 256 + threadIdx.x;   // 0 .. 512*512-1
  const int g = idx >> 9;
  const int h = idx & 511;
  const int t = mult[(size_t)g * GG + inv[h]];
  const float* src = K + (size_t)t * 64;            // [c][d]
  __bf16* dst = kg + (size_t)idx * 64;              // [d][c]
#pragma unroll
  for (int d = 0; d < 8; ++d)
#pragma unroll
    for (int c = 0; c < 8; ++c)
      dst[d * 8 + c] = (__bf16)src[c * 8 + d];
}

// --------------------------------------------------------------------------
// Layer: xout[b][g*8+d] = gelu( scale * sum_k A[b,k]*Bmat[k,(g,d)] )
// grid (64 batch tiles, 8 g-slices) x 256 threads (8 waves).
// Each wave owns a WMMA accumulator for one g-pair (16 output columns).
// --------------------------------------------------------------------------
__global__ __launch_bounds__(256) void layer_kernel(
    const __bf16* __restrict__ xin, const __bf16* __restrict__ kg,
    __bf16* __restrict__ xout) {
  __shared__ __attribute__((aligned(16))) unsigned char Alds[16 * AROW_BYTES];
  const int tid = threadIdx.x;
  const int btile = blockIdx.x;

  // ---- stage 16 x 4096 bf16 A-panel into LDS (16B chunks, 32 iters) ----
  {
    const unsigned char* gsrc =
        (const unsigned char*)(xin + (size_t)btile * 16 * KDIM);
#pragma unroll 4
    for (int it = 0; it < 32; ++it) {
      const int chunk = tid + it * 256;
      const int row = chunk >> 9;            // 0..15
      const int off = (chunk & 511) << 4;    // 0..8176
#if USE_ASYNC
      __builtin_amdgcn_global_load_async_to_lds_b128(
          (v4i*)(gsrc + (size_t)row * (KDIM * 2) + off),
          (v4i*)(Alds + (size_t)row * AROW_BYTES + off),
          0, 0);
#else
      *(v8bf*)(Alds + (size_t)row * AROW_BYTES + off) =
          *(const v8bf*)(gsrc + (size_t)row * (KDIM * 2) + off);
#endif
    }
#if USE_ASYNC
    __builtin_amdgcn_s_wait_asynccnt(0);
#endif
  }
  __syncthreads();

  const int wave = tid >> 5;
  const int lane = tid & 31;
  const int half = lane >> 4;       // which K-half this lane holds
  const int col  = lane & 15;       // A row (M) == B/D column (N)
  const unsigned char* arow = Alds + (size_t)col * AROW_BYTES + half * 16;
  const int gpbase = blockIdx.y * 32;

  for (int gp = gpbase + wave; gp < gpbase + 32; gp += 8) {
    const int g = gp * 2 + (col >> 3);
    const int d = col & 7;
    // column base in Kg: element offset (g*512 + h)*64 + d*8
    const __bf16* bcol = kg + (size_t)g * GG * 64 + (size_t)d * 8;
    v8f acc = {0.f, 0.f, 0.f, 0.f, 0.f, 0.f, 0.f, 0.f};

#pragma unroll 4
    for (int kc = 0; kc < 128; ++kc) {
      // A lane data: K = {half*8 .. +7} and {half*8+16 .. +7} of this chunk
      const unsigned char* ap = arow + kc * 64;
      v8bf alo = *(const v8bf*)ap;
      v8bf ahi = *(const v8bf*)(ap + 32);
      // B lane data: K-half 'half' -> h_local {half, half+2}, c=0..7
      const int h = kc * 4 + half;
      const __bf16* bp = bcol + (size_t)h * 64;
      __builtin_prefetch(bp + 1024, 0, 3);   // pull B stream into near cache
      v8bf blo = *(const v8bf*)bp;
      v8bf bhi = *(const v8bf*)(bp + 128);
      v16bf a, b;
#pragma unroll
      for (int i = 0; i < 8; ++i) {
        a[i] = alo[i]; a[i + 8] = ahi[i];
        b[i] = blo[i]; b[i + 8] = bhi[i];
      }
      acc = __builtin_amdgcn_wmma_f32_16x16x32_bf16(
          false, a, false, b, (short)0, acc, false, false);
    }

    const float scale = 0.04419417382415922f;   // 1/sqrt(512)
    const int brow0 = btile * 16 + half * 8;    // C layout: VGPR r -> M = r + half*8
    __bf16* orow = xout + (size_t)g * 8 + d;
#pragma unroll
    for (int r = 0; r < 8; ++r) {
      float v = gelu_tanh((float)acc[r] * scale);
      orow[(size_t)(brow0 + r) * KDIM] = (__bf16)v;
    }
  }
}

// --------------------------------------------------------------------------
// Head: pooled[d] = mean_g x[b][g*8+d]; out = (pooled@W_amp+b_amp) + i*(...)
// --------------------------------------------------------------------------
__global__ __launch_bounds__(256) void head_kernel(
    const __bf16* __restrict__ x, const float* __restrict__ Wamp,
    const float* __restrict__ bamp, const float* __restrict__ Wph,
    const float* __restrict__ bph, float* __restrict__ out) {
  const int b = blockIdx.x * 256 + threadIdx.x;
  if (b >= NBATCH) return;
  const __bf16* row = x + (size_t)b * KDIM;
  float p[8] = {0.f, 0.f, 0.f, 0.f, 0.f, 0.f, 0.f, 0.f};
  for (int g = 0; g < GG; ++g) {
    v8bf v = *(const v8bf*)(row + g * 8);
#pragma unroll
    for (int d = 0; d < 8; ++d) p[d] += (float)v[d];
  }
  float re = bamp[0], im = bph[0];
#pragma unroll
  for (int d = 0; d < 8; ++d) {
    float pd = p[d] * (1.0f / 512.0f);
    re += pd * Wamp[d];
    im += pd * Wph[d];
  }
  out[2 * b]     = re;   // complex64: interleaved (real, imag)
  out[2 * b + 1] = im;
}

// --------------------------------------------------------------------------
extern "C" void kernel_launch(void* const* d_in, const int* in_sizes, int n_in,
                              void* d_out, int out_size, void* d_ws, size_t ws_size,
                              hipStream_t stream) {
  const float* s      = (const float*)d_in[0];
  const int*   perms  = (const int*)d_in[1];
  const int*   mult   = (const int*)d_in[2];
  const int*   inv    = (const int*)d_in[3];
  const float* W_lift = (const float*)d_in[4];
  const float* b_lift = (const float*)d_in[5];
  const float* K0     = (const float*)d_in[6];
  const float* K1     = (const float*)d_in[7];
  const float* W_amp  = (const float*)d_in[8];
  const float* b_amp  = (const float*)d_in[9];
  const float* W_ph   = (const float*)d_in[10];
  const float* b_ph   = (const float*)d_in[11];
  float* out = (float*)d_out;

  // workspace layout (48MB): xA | xB | Kg
  __bf16* xA = (__bf16*)d_ws;
  __bf16* xB = xA + (size_t)NBATCH * KDIM;                 // +8MB
  __bf16* Kg = xB + (size_t)NBATCH * KDIM;                 // +8MB, Kg = 33.5MB

  lift_kernel<<<dim3(512, 4), 256, 0, stream>>>(s, perms, W_lift, b_lift, xA);

  pack_kernel<<<1024, 256, 0, stream>>>(K0, mult, inv, Kg);
  layer_kernel<<<dim3(64, 8), 256, 0, stream>>>(xA, Kg, xB);

  pack_kernel<<<1024, 256, 0, stream>>>(K1, mult, inv, Kg);
  layer_kernel<<<dim3(64, 8), 256, 0, stream>>>(xB, Kg, xA);

  head_kernel<<<4, 256, 0, stream>>>(xA, W_amp, b_amp, W_ph, b_ph, out);
}
